// DicGaussianRBF_31482110280409
// MI455X (gfx1250) — compile-verified, matrix operand order checked by reading, not run on hardware
//
#include <hip/hip_runtime.h>
#include <hip/hip_bf16.h>

// CDNA5 / gfx1250: wave32, WMMA 16x16x32 bf16 -> f32 accumulate.

typedef __attribute__((ext_vector_type(16))) __bf16        v16bf;
typedef __attribute__((ext_vector_type(8)))  float         v8f;
typedef __attribute__((ext_vector_type(4)))  unsigned int  v4u;

#define DDIM      256
#define KDIM      2048
#define OUTW      2305          // 1 + 256 + 2048
#define TILE      128
#define KSTEP     32
#define NSTEPS    (DDIM / KSTEP)
#define ROWBYTES  80            // 64B of bf16 K-data + 16B pad (bank spread)

// f32 pair -> packed bf16 via the hardware converter (v_cvt_pk_bf16_f32)
__device__ __forceinline__ unsigned int pack2_bf16(float a, float b) {
  union { __bf16 h[2]; unsigned int u; } r;
  r.h[0] = static_cast<__bf16>(a);
  r.h[1] = static_cast<__bf16>(b);
  return r.u;
}

// Per-lane fragment fetch following the CDNA5 16-bit A/B layout:
// lanes 0-15 hold K=0..7 (bytes [0,16)) and K=16..23 (bytes [32,48)),
// lanes 16-31 hold K=8..15 (bytes [16,32)) and K=24..31 (bytes [48,64)).
__device__ __forceinline__ v16bf load_frag(const unsigned char* base, int rowBase, int lane) {
  const int row  = rowBase + (lane & 15);
  const int half = lane >> 4;
  const unsigned char* p = base + row * ROWBYTES + half * 16;
  union { v4u q[2]; v16bf v; } u;
  u.q[0] = *(const v4u*)(p);
  u.q[1] = *(const v4u*)(p + 32);
  return u.v;
}

__global__ void __launch_bounds__(256)
rbf_gemm_kernel(const float* __restrict__ data,
                const float* __restrict__ centers,
                float* __restrict__ out) {
  __shared__ __align__(16) unsigned char smA[2][TILE * ROWBYTES];
  __shared__ __align__(16) unsigned char smB[2][TILE * ROWBYTES];
  __shared__ float s_x2[TILE];
  __shared__ float s_c2[TILE];

  const int tid  = threadIdx.x;
  const int lane = tid & 31;
  const int wid  = tid >> 5;
  const int wr   = wid & 1;      // wave row group: 2 x 64 rows
  const int wc   = wid >> 1;     // wave col group: 4 x 32 cols
  const int rowBlk = blockIdx.y * TILE;
  const int colBlk = blockIdx.x * TILE;

  // ---- staging assignment (wave-uniform split; EXEC stays all-1s) ----
  // waves 0..3 (tid 0..127)   : stage A row  (rowBlk + tid),     full 32-K slice
  // waves 4..7 (tid 128..255) : stage B row  (colBlk + tid-128), full 32-K slice
  const int  srow = tid & 127;
  const bool isA  = tid < 128;
  const float4* gsrc = (const float4*)((isA ? data + (size_t)(rowBlk + srow) * DDIM
                                            : centers + (size_t)(colBlk + srow) * DDIM));
  unsigned char* ldsRow0 = (isA ? &smA[0][srow * ROWBYTES] : &smB[0][srow * ROWBYTES]);
  unsigned char* ldsRow1 = (isA ? &smA[1][srow * ROWBYTES] : &smB[1][srow * ROWBYTES]);

  float  sum = 0.f;             // ||row||^2 accumulated for free during staging
  float4 rr[2][8];              // two in-flight 32-float K-slices (register pipeline)

  auto issue_loads = [&](int slot, int k) {
    const float4* p = gsrc + k * (KSTEP / 4);
#pragma unroll
    for (int q = 0; q < 8; ++q) rr[slot][q] = p[q];
  };

  auto cvt_store = [&](int slot, int buf) {
    const float* f = (const float*)rr[slot];
    unsigned int pk[16];
    float s = 0.f;
#pragma unroll
    for (int i = 0; i < 16; ++i) {
      float a0 = f[2 * i], a1 = f[2 * i + 1];
      s += a0 * a0 + a1 * a1;
      pk[i] = pack2_bf16(a0, a1);
    }
    sum += s;
    v4u* d = (v4u*)(buf ? ldsRow1 : ldsRow0);
#pragma unroll
    for (int q = 0; q < 4; ++q) {
      v4u t;
      t[0] = pk[4 * q + 0]; t[1] = pk[4 * q + 1];
      t[2] = pk[4 * q + 2]; t[3] = pk[4 * q + 3];
      d[q] = t;
    }
  };

  v8f acc[4][2] = {};

  // ---- prologue: stage step 0 into LDS buf0; put step 1 loads in flight ----
  issue_loads(0, 0);
  issue_loads(1, 1);            // in flight across cvt_store(0) + compute(0)
  cvt_store(0, 0);              // waits only for slot-0 loads (loadcnt <= 8)
  __syncthreads();

  // ---- main loop: step k+2 loads issued under step k's WMMAs ----
#pragma unroll
  for (int k = 0; k < NSTEPS; ++k) {
    const int buf = k & 1;

    v16bf af[4], bfg[2];
#pragma unroll
    for (int i = 0; i < 4; ++i) af[i]  = load_frag(smA[buf], wr * 64 + i * 16, lane);
#pragma unroll
    for (int j = 0; j < 2; ++j) bfg[j] = load_frag(smB[buf], wc * 32 + j * 16, lane);
#pragma unroll
    for (int i = 0; i < 4; ++i)
#pragma unroll
      for (int j = 0; j < 2; ++j)
        acc[i][j] = __builtin_amdgcn_wmma_f32_16x16x32_bf16(
            /*neg_a=*/false, af[i], /*neg_b=*/false, bfg[j],
            /*c_mod=*/(short)0, acc[i][j], /*reuse_a=*/false, /*reuse_b=*/false);

    if (k + 1 < NSTEPS) {
      // issue loads for step k+2 now: no LDS touched, so legal before the
      // barrier; consumed only at the end of iteration k+1 -> a full compute
      // phase of latency hiding.
      if (k + 2 < NSTEPS) issue_loads(k & 1, k + 2);
      __syncthreads();                       // all waves done reading buf^1
      cvt_store((k + 1) & 1, buf ^ 1);       // slot (k+1)&1 holds step k+1
      __syncthreads();                       // stores visible before next compute
    }
  }

  // ---- squared norms: each staging thread owns one full row ----
  if (isA) s_x2[srow] = sum;
  else     s_c2[srow] = sum;
  __syncthreads();

  // ---- epilogue: r^2 = x2 + c2 - 2*xc ; out = exp(-5 r^2) ----
  // C/D layout: VGPR v -> M = v (lanes 0-15) or v+8 (lanes 16-31), N = lane%16
  const int mhi = (lane >> 4) * 8;
  const int nlo = lane & 15;
#pragma unroll
  for (int i = 0; i < 4; ++i) {
#pragma unroll
    for (int j = 0; j < 2; ++j) {
      const int cl   = wc * 32 + j * 16 + nlo;
      const float c2 = s_c2[cl];
      const size_t ocol = (size_t)(colBlk + cl) + 1 + DDIM;
#pragma unroll
      for (int v = 0; v < 8; ++v) {
        const int rl = wr * 64 + i * 16 + mhi + v;
        const float r2 = s_x2[rl] + c2 - 2.0f * acc[i][j][v];
        out[(size_t)(rowBlk + rl) * OUTW + ocol] = __expf(-5.0f * r2);
      }
    }
  }
}

// ones column + verbatim data copy (columns 0 .. 256)
__global__ void __launch_bounds__(256)
rbf_head_kernel(const float* __restrict__ data, float* __restrict__ out) {
  const size_t n = blockIdx.x;
  const int d = threadIdx.x;
  const float v = data[n * DDIM + d];
  float* o = out + n * OUTW;
  o[1 + d] = v;
  if (d == 0) o[0] = 1.0f;
}

extern "C" void kernel_launch(void* const* d_in, const int* in_sizes, int n_in,
                              void* d_out, int out_size, void* d_ws, size_t ws_size,
                              hipStream_t stream) {
  const float* data    = (const float*)d_in[0];   // [N, 256]
  const float* centers = (const float*)d_in[1];   // [K, 256]
  float* out = (float*)d_out;                     // [N, 2305]

  const int N = in_sizes[0] / DDIM;               // 65536
  const int K = in_sizes[1] / DDIM;               // 2048

  dim3 grid(K / TILE, N / TILE);                  // x = col blocks -> L2 reuse of data band
  rbf_gemm_kernel<<<grid, dim3(256), 0, stream>>>(data, centers, out);
  rbf_head_kernel<<<dim3(N), dim3(256), 0, stream>>>(data, out);
}
